// WSDDN_76295799046515
// MI455X (gfx1250) — compile-verified, matrix operand order checked by reading, not run on hardware
//
#include <hip/hip_runtime.h>
#include <float.h>
#include <math.h>
#include <stdint.h>

// ---------------------------------------------------------------------------
// WSDDN forward loss on MI455X (gfx1250, wave32, WMMA).
// Hot path: 2x GEMM 256x4096x4096 with v_wmma_f32_16x16x32_f16.
//   - A tile (16 x 4096 f16 = 128KB) staged in LDS via async global->LDS copies
//     (ASYNCcnt path), shared by all 8 waves of the block.
//   - Each wave owns 4 N-tiles: per K-step 2 ds_load_b128 + 8 global_load_b128
//     feed 4 WMMA.
// ---------------------------------------------------------------------------

typedef _Float16 v8h  __attribute__((ext_vector_type(8)));
typedef _Float16 v16h __attribute__((ext_vector_type(16)));
typedef float    v8f  __attribute__((ext_vector_type(8)));

#define C_CH   512
#define HFM    50
#define WFM    50
#define R_ROIS 256
#define NCLS   20
#define FMX    12
#define ROISCL 16
#define DF     4096   // feature dim = C_CH * 8
#define NT     4      // N-tiles (16 cols each) per wave

// ---------------------------------------------------------------------------
// 1) ROI adaptive max pooling: feats = 0.5*(pool2x4 + pool4x2), flattened.
// ---------------------------------------------------------------------------
__device__ __forceinline__ float binmax(const float* __restrict__ fm,
                                        int c, int x, int y, int w, int h,
                                        int nbx, int nby, int bx, int by) {
  int xs = bx * w / nbx;
  int xe = ((bx + 1) * w + nbx - 1) / nbx; if (xe > w) xe = w;
  int ys = by * h / nby;
  int ye = ((by + 1) * h + nby - 1) / nby; if (ye > h) ye = h;
  const float* base = fm + (size_t)c * (HFM * WFM);
  float m = -FLT_MAX;
  for (int i = xs; i < xe; ++i)
    for (int j = ys; j < ye; ++j)
      m = fmaxf(m, base[(x + i) * WFM + (y + j)]);
  return m;
}

__global__ void __launch_bounds__(256)
roi_pool_kernel(const float* __restrict__ fm, const int* __restrict__ rois,
                float* __restrict__ featsF, _Float16* __restrict__ featsH) {
  int idx = blockIdx.x * blockDim.x + threadIdx.x;
  if (idx >= R_ROIS * DF) return;
  int r = idx >> 12;        // /4096
  int k = idx & (DF - 1);
  int c = k >> 3;
  int p = k & 7;
  int x = rois[r * 4 + 0] / ROISCL;
  int y = rois[r * 4 + 1] / ROISCL;
  int w = rois[r * 4 + 2] / ROISCL;
  int h = rois[r * 4 + 3] / ROISCL;
  float v24 = binmax(fm, c, x, y, w, h, 2, 4, p >> 2, p & 3);
  float v42 = binmax(fm, c, x, y, w, h, 4, 2, p >> 1, p & 1);
  float v = 0.5f * (v24 + v42);
  featsF[idx] = v;
  featsH[idx] = (_Float16)v;
}

// ---------------------------------------------------------------------------
// 2) Transpose 4096x4096 fp32 -> f16 (Wt[n][k] = W[k][n]) via LDS tiles.
// ---------------------------------------------------------------------------
__global__ void __launch_bounds__(256)
transpose_w_kernel(const float* __restrict__ W, _Float16* __restrict__ Wt) {
  __shared__ float tile[32][33];
  int bx = blockIdx.x * 32;   // n base
  int by = blockIdx.y * 32;   // k base
  int tx = threadIdx.x;       // 0..31
  int ty = threadIdx.y;       // 0..7
  #pragma unroll
  for (int i = 0; i < 32; i += 8)
    tile[ty + i][tx] = W[(size_t)(by + ty + i) * DF + (bx + tx)];
  __syncthreads();
  #pragma unroll
  for (int i = 0; i < 32; i += 8)
    Wt[(size_t)(bx + ty + i) * DF + (by + tx)] = (_Float16)tile[tx][ty + i];
}

// ---------------------------------------------------------------------------
// 3) WMMA GEMM:  C[M=256,N=4096] = A[256,4096] * B + bias
//    A row-major f16 (staged in LDS); B given as Bt[n][k] (column-major) f16.
//    Grid: (M/16, N/(16*8*NT)); block = 8 waves; wave owns NT n-tiles.
//    Fragment layout (16-bit A 16x32): lane<16 -> K {0..7,16..23},
//    lane>=16 -> K {8..15,24..31}; row/col = lane%16.
// ---------------------------------------------------------------------------
__global__ void __launch_bounds__(256)
wmma_gemm_kernel(const _Float16* __restrict__ A,
                 const _Float16* __restrict__ Bt,
                 const float* __restrict__ bias,
                 float* __restrict__ Cf,
                 _Float16* __restrict__ Ch) {
  __shared__ _Float16 Als[16 * DF];       // 128 KB A tile
  const int K = DF;
  int tid   = threadIdx.x;
  int wave  = tid >> 5;                   // 0..7
  int lane  = tid & 31;
  int mTile = blockIdx.x;                 // 0..15  (M/16)
  int nBase = (blockIdx.y * 8 + wave) * NT; // first of NT n-tiles for this wave
  int half  = lane >> 4;                  // 0/1
  int lm    = lane & 15;

  // ---- stage A tile (16 x 4096 f16) into LDS with async global->LDS copies.
  // NOTE: the LDS address operand is derived from the Als pointer itself
  // (flat shared address low 32 bits == LDS byte address per the aperture
  // rules). The ptrtoint also makes Als escape, so the "memory" clobber
  // correctly forces later ds_load reads to see the async-written data.
  {
    uint64_t gbase = (uint64_t)(uintptr_t)(A + (size_t)mTile * 16 * DF);
    #pragma unroll 4
    for (int i = tid; i < (16 * DF) / 8; i += 256) {   // 16B chunks
      unsigned ldsOff = (unsigned)(uintptr_t)(Als + (size_t)i * 8);
      uint64_t gaddr  = gbase + (uint64_t)i * 16u;
      asm volatile("global_load_async_to_lds_b128 %0, %1, off"
                   :: "v"(ldsOff), "v"(gaddr) : "memory");
    }
    asm volatile("s_wait_asynccnt 0x0" ::: "memory");
  }
  __syncthreads();

  const _Float16* aLds = Als + (size_t)lm * K + half * 8;
  const _Float16* bRow0 = Bt + (size_t)((nBase + 0) * 16 + lm) * K + half * 8;
  const _Float16* bRow1 = Bt + (size_t)((nBase + 1) * 16 + lm) * K + half * 8;
  const _Float16* bRow2 = Bt + (size_t)((nBase + 2) * 16 + lm) * K + half * 8;
  const _Float16* bRow3 = Bt + (size_t)((nBase + 3) * 16 + lm) * K + half * 8;

  v8f acc0 = {}, acc1 = {}, acc2 = {}, acc3 = {};
  #pragma unroll 2
  for (int k0 = 0; k0 < K; k0 += 32) {
    if (k0 + 512 < K) {
      __builtin_prefetch(bRow0 + k0 + 512, 0, 1);
      __builtin_prefetch(bRow1 + k0 + 512, 0, 1);
      __builtin_prefetch(bRow2 + k0 + 512, 0, 1);
      __builtin_prefetch(bRow3 + k0 + 512, 0, 1);
    }
    v8h a0 = *(const v8h*)(aLds + k0);
    v8h a1 = *(const v8h*)(aLds + k0 + 16);
    v16h a = __builtin_shufflevector(a0, a1, 0,1,2,3,4,5,6,7,8,9,10,11,12,13,14,15);

    v8h p0 = *(const v8h*)(bRow0 + k0);
    v8h p1 = *(const v8h*)(bRow0 + k0 + 16);
    v16h b = __builtin_shufflevector(p0, p1, 0,1,2,3,4,5,6,7,8,9,10,11,12,13,14,15);
    acc0 = __builtin_amdgcn_wmma_f32_16x16x32_f16(false, a, false, b, (short)0, acc0, false, false);

    p0 = *(const v8h*)(bRow1 + k0);
    p1 = *(const v8h*)(bRow1 + k0 + 16);
    b  = __builtin_shufflevector(p0, p1, 0,1,2,3,4,5,6,7,8,9,10,11,12,13,14,15);
    acc1 = __builtin_amdgcn_wmma_f32_16x16x32_f16(false, a, false, b, (short)0, acc1, false, false);

    p0 = *(const v8h*)(bRow2 + k0);
    p1 = *(const v8h*)(bRow2 + k0 + 16);
    b  = __builtin_shufflevector(p0, p1, 0,1,2,3,4,5,6,7,8,9,10,11,12,13,14,15);
    acc2 = __builtin_amdgcn_wmma_f32_16x16x32_f16(false, a, false, b, (short)0, acc2, false, false);

    p0 = *(const v8h*)(bRow3 + k0);
    p1 = *(const v8h*)(bRow3 + k0 + 16);
    b  = __builtin_shufflevector(p0, p1, 0,1,2,3,4,5,6,7,8,9,10,11,12,13,14,15);
    acc3 = __builtin_amdgcn_wmma_f32_16x16x32_f16(false, a, false, b, (short)0, acc3, false, false);
  }

  // ---- writeback: C layout: VGPR j -> M = half*8 + j; N = lane%16
  v8f accs[NT] = {acc0, acc1, acc2, acc3};
  #pragma unroll
  for (int t = 0; t < NT; ++t) {
    int col = (nBase + t) * 16 + lm;
    float bv = bias[col];
    #pragma unroll
    for (int j = 0; j < 8; ++j) {
      int row = mTile * 16 + half * 8 + j;
      float v = accs[t][j] + bv;
      Cf[(size_t)row * DF + col] = v;
      Ch[(size_t)row * DF + col] = (_Float16)v;
    }
  }
}

// ---------------------------------------------------------------------------
// 4) Head logits: logits[r][c] = fc7[r] . W8[:,c] + b8[c]   (4096 x 20)
// ---------------------------------------------------------------------------
__global__ void __launch_bounds__(256)
head_logits_kernel(const float* __restrict__ fc7, const float* __restrict__ W8,
                   const float* __restrict__ b8, float* __restrict__ logits) {
  __shared__ float red[NCLS];
  int r = blockIdx.x;
  int t = threadIdx.x;
  float acc[NCLS];
  #pragma unroll
  for (int c = 0; c < NCLS; ++c) acc[c] = 0.f;
  for (int k = t; k < DF; k += 256) {
    float f = fc7[(size_t)r * DF + k];
    const float* wr = W8 + (size_t)k * NCLS;
    #pragma unroll
    for (int c = 0; c < NCLS; ++c) acc[c] += f * wr[c];
  }
  if (t < NCLS) red[t] = 0.f;
  __syncthreads();
  #pragma unroll
  for (int c = 0; c < NCLS; ++c) atomicAdd(&red[c], acc[c]);
  __syncthreads();
  if (t < NCLS) logits[r * NCLS + t] = red[t] + b8[t];
}

// ---------------------------------------------------------------------------
// 5) Softmaxes, scores, out-vector, BCE.  Single block of 256 threads.
// ---------------------------------------------------------------------------
__global__ void __launch_bounds__(256)
score_bce_kernel(const float* __restrict__ lc, const float* __restrict__ ld,
                 const float* __restrict__ label,
                 float* __restrict__ scores, float* __restrict__ dout) {
  __shared__ float colmax[NCLS], colsum[NCLS], outsh[NCLS];
  int t = threadIdx.x;
  if (t < NCLS) {
    float m = -FLT_MAX;
    for (int r = 0; r < R_ROIS; ++r) m = fmaxf(m, ld[r * NCLS + t]);
    float s = 0.f;
    for (int r = 0; r < R_ROIS; ++r) s += expf(ld[r * NCLS + t] - m);
    colmax[t] = m; colsum[t] = s; outsh[t] = 0.f;
  }
  __syncthreads();
  float rm = -FLT_MAX;
  #pragma unroll
  for (int c = 0; c < NCLS; ++c) rm = fmaxf(rm, lc[t * NCLS + c]);
  float rs = 0.f;
  #pragma unroll
  for (int c = 0; c < NCLS; ++c) rs += expf(lc[t * NCLS + c] - rm);
  #pragma unroll
  for (int c = 0; c < NCLS; ++c) {
    float pc = expf(lc[t * NCLS + c] - rm) / rs;
    float pd = expf(ld[t * NCLS + c] - colmax[c]) / colsum[c];
    float sc = pc * pd;
    scores[t * NCLS + c] = sc;
    atomicAdd(&outsh[c], sc);
  }
  __syncthreads();
  if (t == 0) {
    float bce = 0.f;
    for (int c = 0; c < NCLS; ++c) {
      float o = outsh[c];
      o = fminf(fmaxf(o, 1e-7f), 1.f - 1e-7f);
      bce += -(label[c] * logf(o) + (1.f - label[c]) * log1pf(-o));
    }
    dout[0] = bce / (float)NCLS;   // initializes output
  }
}

// ---------------------------------------------------------------------------
// 6) Per-class regularizer; one block per class, atomicAdd into d_out[0].
// ---------------------------------------------------------------------------
__global__ void __launch_bounds__(256)
reg_kernel(const float* __restrict__ scores, const float* __restrict__ fc7,
           const int* __restrict__ rois, const float* __restrict__ label,
           float* __restrict__ dout) {
  int c = blockIdx.x;
  float lab = label[c];
  if (lab == 0.f) return;  // uniform per block

  __shared__ float sk[R_ROIS];
  __shared__ float msk[R_ROIS];
  __shared__ int   topIdx;
  __shared__ float rbuf[256];
  int t = threadIdx.x;

  sk[t] = scores[t * NCLS + c];
  __syncthreads();
  if (t == 0) {
    float m = sk[0]; int mi = 0;
    for (int r = 1; r < R_ROIS; ++r) { if (sk[r] < m) { m = sk[r]; mi = r; } }
    topIdx = mi;
  }
  __syncthreads();
  int top = topIdx;

  {
    float tx = (float)(rois[top * 4 + 0] / ROISCL);
    float ty = (float)(rois[top * 4 + 1] / ROISCL);
    float tw = (float)(rois[top * 4 + 2] / ROISCL);
    float th = (float)(rois[top * 4 + 3] / ROISCL);
    float bx = (float)(rois[t * 4 + 0] / ROISCL);
    float by = (float)(rois[t * 4 + 1] / ROISCL);
    float bw = (float)(rois[t * 4 + 2] / ROISCL);
    float bh = (float)(rois[t * 4 + 3] / ROISCL);
    float ix1 = fmaxf(bx, tx), iy1 = fmaxf(by, ty);
    float ix2 = fminf(bx + bw, tx + tw), iy2 = fminf(by + bh, ty + th);
    float inter = fmaxf(ix2 - ix1 + 1.f, 0.f) * fmaxf(iy2 - iy1 + 1.f, 0.f);
    float a1 = (bw + 1.f) * (bh + 1.f);
    float a2 = (tw + 1.f) * (th + 1.f);
    float iou = inter / (a1 + a2 - inter);
    msk[t] = (iou > 0.6f) ? 1.f : 0.f;
  }
  __syncthreads();

  float ftop[DF / 256];
  #pragma unroll
  for (int i = 0; i < DF / 256; ++i)
    ftop[i] = fc7[(size_t)top * DF + t + i * 256];

  float acc = 0.f;
  for (int r = 0; r < R_ROIS; ++r) {
    if (r == top) continue;
    float w = sk[r], m = msk[r];
    #pragma unroll
    for (int i = 0; i < DF / 256; ++i) {
      float diff = (fc7[(size_t)r * DF + t + i * 256] * m - ftop[i]) * w;
      acc += diff * diff;
    }
  }
  rbuf[t] = acc;
  __syncthreads();
  for (int s = 128; s > 0; s >>= 1) {
    if (t < s) rbuf[t] += rbuf[t + s];
    __syncthreads();
  }
  if (t == 0) atomicAdd(dout, 0.5f * rbuf[0] * lab / (float)NCLS);
}

// ---------------------------------------------------------------------------
// Launcher
// ---------------------------------------------------------------------------
extern "C" void kernel_launch(void* const* d_in, const int* in_sizes, int n_in,
                              void* d_out, int out_size, void* d_ws, size_t ws_size,
                              hipStream_t stream) {
  const float* fm   = (const float*)d_in[0];
  const int*   rois = (const int*)  d_in[1];
  const float* lab  = (const float*)d_in[2];
  const float* W6   = (const float*)d_in[3];
  const float* b6   = (const float*)d_in[4];
  const float* W7   = (const float*)d_in[5];
  const float* b7   = (const float*)d_in[6];
  const float* W8c  = (const float*)d_in[7];
  const float* b8c  = (const float*)d_in[8];
  const float* W8d  = (const float*)d_in[9];
  const float* b8d  = (const float*)d_in[10];
  float* out = (float*)d_out;
  char* ws = (char*)d_ws;

  size_t off = 0;
  auto take = [&](size_t bytes) { size_t o = off; off += (bytes + 255) & ~(size_t)255; return o; };

  float*    featsF  = (float*)   (ws + take((size_t)R_ROIS * DF * 4));
  _Float16* featsH  = (_Float16*)(ws + take((size_t)R_ROIS * DF * 2));
  _Float16* W6t     = (_Float16*)(ws + take((size_t)DF * DF * 2));
  _Float16* W7t     = (_Float16*)(ws + take((size_t)DF * DF * 2));
  float*    fc6F    = (float*)   (ws + take((size_t)R_ROIS * DF * 4));
  _Float16* fc6H    = (_Float16*)(ws + take((size_t)R_ROIS * DF * 2));
  float*    fc7F    = (float*)   (ws + take((size_t)R_ROIS * DF * 4));
  _Float16* fc7H    = (_Float16*)(ws + take((size_t)R_ROIS * DF * 2));
  float*    logitsC = (float*)   (ws + take((size_t)R_ROIS * NCLS * 4));
  float*    logitsD = (float*)   (ws + take((size_t)R_ROIS * NCLS * 4));
  float*    scores  = (float*)   (ws + take((size_t)R_ROIS * NCLS * 4));

  // 1) ROI pooling -> feats (fp32 + f16)
  roi_pool_kernel<<<(R_ROIS * DF) / 256, 256, 0, stream>>>(fm, rois, featsF, featsH);

  // 2) Weight downconvert + transpose
  {
    dim3 grid(DF / 32, DF / 32), blk(32, 8);
    transpose_w_kernel<<<grid, blk, 0, stream>>>(W6, W6t);
    transpose_w_kernel<<<grid, blk, 0, stream>>>(W7, W7t);
  }

  // 3) fc6 = feats @ W6 + b6 ; fc7 = fc6 @ W7 + b7   (WMMA, async-LDS A tile)
  {
    dim3 grid(R_ROIS / 16, DF / (16 * 8 * NT));   // (16, 8)
    wmma_gemm_kernel<<<grid, 256, 0, stream>>>(featsH, W6t, b6, fc6F, fc6H);
    wmma_gemm_kernel<<<grid, 256, 0, stream>>>(fc6H,  W7t, b7, fc7F, fc7H);
  }

  // 4) heads
  head_logits_kernel<<<R_ROIS, 256, 0, stream>>>(fc7F, W8c, b8c, logitsC);
  head_logits_kernel<<<R_ROIS, 256, 0, stream>>>(fc7F, W8d, b8d, logitsD);

  // 5) softmax / scores / bce -> d_out[0]
  score_bce_kernel<<<1, 256, 0, stream>>>(logitsC, logitsD, lab, scores, out);

  // 6) regularizer, accumulated into d_out[0]
  reg_kernel<<<NCLS, 256, 0, stream>>>(scores, fc7F, rois, lab, out);
}
